// GraphTransformer_41120016892388
// MI455X (gfx1250) — compile-verified, matrix operand order checked by reading, not run on hardware
//
#include <hip/hip_runtime.h>
#include <hip/hip_bf16.h>

// Problem constants (match reference)
#define NN 20000
#define EE 640000
#define DD 256
#define HH 4
#define LL 3

typedef __bf16 bf16;
typedef __attribute__((ext_vector_type(16))) __bf16 v16bf;
typedef __attribute__((ext_vector_type(8)))  float  v8f;
typedef __attribute__((ext_vector_type(4)))  int    v4i;

// ---- CDNA5 async global->LDS path (builtin confirmed present by probe) ----
// Signature (from hipcc diagnostic): param0 = v4i addrspace(1)*, 4 args total.
#if __has_builtin(__builtin_amdgcn_global_load_async_to_lds_b128)
  #define USE_ASYNC_LDS 1
  #define ASYNC_FILL(g, l) __builtin_amdgcn_global_load_async_to_lds_b128(          \
      (__attribute__((address_space(1))) v4i*)(g),                                  \
      (__attribute__((address_space(3))) v4i*)(l), 0, 0)
  #if __has_builtin(__builtin_amdgcn_s_wait_asynccnt)
    #define ASYNC_WAIT() __builtin_amdgcn_s_wait_asynccnt(0)
  #else
    #define ASYNC_WAIT() asm volatile("s_wait_asynccnt 0" ::: "memory")
  #endif
#endif

// ---------------- ordered-uint float max encoding ----------------
__device__ __forceinline__ unsigned f2ord(float x) {
  unsigned b = __float_as_uint(x);
  return (b & 0x80000000u) ? ~b : (b | 0x80000000u);
}
__device__ __forceinline__ float ord2f(unsigned u) {
  return __uint_as_float((u & 0x80000000u) ? (u & 0x7FFFFFFFu) : ~u);
}

// ---------------- small utility kernels ----------------
__global__ __launch_bounds__(256) void zero_u32(unsigned* __restrict__ p, int n) {
  int t = blockIdx.x * 256 + threadIdx.x;
  if (t < n) p[t] = 0u;
}

__global__ __launch_bounds__(256) void cvt_x_bf16(const float* __restrict__ x,
                                                  bf16* __restrict__ xb) {
  int t = blockIdx.x * 256 + threadIdx.x;   // N*D threads
  xb[t] = (bf16)x[t];
}

// WbT[m][n][k] = W_m[k][n]  (transpose so B-fragment K-pairs are contiguous)
__global__ __launch_bounds__(256) void cvt_w_bf16(const float* __restrict__ Wq,
                                                  const float* __restrict__ Wk,
                                                  const float* __restrict__ Wv,
                                                  const float* __restrict__ Ws,
                                                  bf16* __restrict__ WbT) {
  int t = blockIdx.x * 256 + threadIdx.x;   // 4*256*256 threads
  int m   = t >> 16;
  int rem = t & 65535;
  int n   = rem >> 8;
  int k   = rem & 255;
  const float* W = (m == 0) ? Wq : (m == 1) ? Wk : (m == 2) ? Wv : Ws;
  WbT[t] = (bf16)W[k * DD + n];
}

// ---------------- fused 4-matrix WMMA GEMM with LDS-staged weights ----------------
// C_m = Xb[20000x256](bf16) @ W_m[256x256](bf16,T) + b_m   -> f32
// Block (256 thr = 8 waves) owns one (matrix m, 64-col group cg) weight panel,
// staged once in LDS (async if available), and 8 consecutive 16-row tiles.
// Grid: (157 row groups, 4 col groups, 4 matrices); tail waves idle past row 19999.
#define LDS_STRIDE 528   // bytes per column row: 512 data + 16 pad (bank spread)

__global__ __launch_bounds__(256) void gemm4_wmma(
    const bf16* __restrict__ Xb, const bf16* __restrict__ WbT,
    const float* __restrict__ bq, const float* __restrict__ bk,
    const float* __restrict__ bv, const float* __restrict__ bs,
    float* __restrict__ Q, float* __restrict__ K,
    float* __restrict__ V, float* __restrict__ S) {
  __shared__ __align__(16) char ldsB[64 * LDS_STRIDE];   // 33,792 B

  const int m    = blockIdx.z;
  const int cg   = blockIdx.y;
  const int wave = threadIdx.x >> 5;
  const int lane = threadIdx.x & 31;
  const int rt   = blockIdx.x * 8 + wave;                // 16-row tile index

  // ---- cooperative fill of the 64-col x 256-K bf16 weight panel ----
  const char* panel =
      (const char*)(WbT + ((size_t)m * DD + (size_t)cg * 64) * DD);
#pragma unroll
  for (int it = 0; it < 8; ++it) {
    const int ch  = it * 256 + threadIdx.x;              // 2048 chunks of 16B
    const int col = ch >> 5;
    const int cp  = ch & 31;
    const char* g = panel + col * 512 + cp * 16;
    char*       l = ldsB + col * LDS_STRIDE + cp * 16;
#if defined(USE_ASYNC_LDS)
    ASYNC_FILL(g, l);
#else
    *(uint4*)l = *(const uint4*)g;
#endif
  }
#if defined(USE_ASYNC_LDS)
  ASYNC_WAIT();
#endif
  __syncthreads();

  if (rt < 1250) {
    float* outp; const float* bias;
    if      (m == 0) { outp = Q; bias = bq; }
    else if (m == 1) { outp = K; bias = bk; }
    else if (m == 2) { outp = V; bias = bv; }
    else             { outp = S; bias = bs; }

    const int r = lane & 15;          // M (A) / N (B,C,D)
    const int h = lane >> 4;          // lane half

    const char* Arow = (const char*)(Xb + (size_t)(rt * 16 + r) * DD);
    __builtin_prefetch(Arow, 0, 1);

    v8f acc[4] = {};
    union Frag { v16bf v; uint4 q[2]; };

#pragma unroll
    for (int ks = 0; ks < DD; ks += 32) {
      // A fragment: 16-bit A layout pairs are two contiguous 16B runs
      Frag a;
      const char* ab = Arow + ks * 2 + h * 16;
      a.q[0] = *(const uint4*)ab;
      a.q[1] = *(const uint4*)(ab + 32);
#pragma unroll
      for (int t = 0; t < 4; ++t) {
        const int ct = t * 16 + r;    // panel-local column
        Frag b;
        const char* lb = ldsB + ct * LDS_STRIDE + ks * 2 + h * 16;
        b.q[0] = *(const uint4*)lb;
        b.q[1] = *(const uint4*)(lb + 32);
        acc[t] = __builtin_amdgcn_wmma_f32_16x16x32_bf16(
            false, a.v, false, b.v, (short)0, acc[t], false, false);
      }
    }

#pragma unroll
    for (int t = 0; t < 4; ++t) {
      const int col  = cg * 64 + t * 16 + r;
      const float bc = bias[col];
#pragma unroll
      for (int g = 0; g < 8; ++g) {
        const int row = rt * 16 + g + 8 * h;
        outp[(size_t)row * DD + col] = acc[t][g] + bc;
      }
    }
  }
}

// ---------------- pass A: per-edge per-head attention scores + segment max ----------------
// One wave per edge; lane l covers feature floats [8l, 8l+8) -> head = l>>3.
__global__ __launch_bounds__(256) void edge_scores(
    const int* __restrict__ ei, const float* __restrict__ Q,
    const float* __restrict__ Kf, float* __restrict__ score,
    unsigned* __restrict__ nodeMax) {
  const int wid = (int)((blockIdx.x * 256 + threadIdx.x) >> 5);
  if (wid >= EE) return;
  const int lane = threadIdx.x & 31;
  const int src = ei[wid];
  const int dst = ei[EE + wid];
  const float4* qp = (const float4*)(Q  + (size_t)dst * DD + lane * 8);
  const float4* kp = (const float4*)(Kf + (size_t)src * DD + lane * 8);
  float4 q0 = qp[0], q1 = qp[1];
  float4 k0 = kp[0], k1 = kp[1];
  float s = q0.x * k0.x + q0.y * k0.y + q0.z * k0.z + q0.w * k0.w +
            q1.x * k1.x + q1.y * k1.y + q1.z * k1.z + q1.w * k1.w;
  s += __shfl_xor(s, 1, 32);
  s += __shfl_xor(s, 2, 32);
  s += __shfl_xor(s, 4, 32);
  if ((lane & 7) == 0) {
    const int hh = lane >> 3;
    const float sc = s * 0.125f;          // 1/sqrt(64)
    score[(size_t)wid * HH + hh] = sc;
    atomicMax(nodeMax + dst * HH + hh, f2ord(sc));
  }
}

// ---------------- pass B: exp(score - max) and segment-sum denominators ----------------
__global__ __launch_bounds__(256) void edge_exp(
    const int* __restrict__ ei, float* __restrict__ score,
    const unsigned* __restrict__ nodeMax, float* __restrict__ denom) {
  const int t = blockIdx.x * 256 + threadIdx.x;   // E*H threads
  if (t >= EE * HH) return;
  const int e  = t >> 2;
  const int hh = t & 3;
  const int dst = ei[EE + e];
  const float mv = ord2f(nodeMax[dst * HH + hh]);
  const float ex = __expf(score[t] - mv);
  score[t] = ex;
  atomicAdd(denom + dst * HH + hh, ex);
}

// ---------------- pass C: alpha * v[src] scattered into per-dst accumulator ----------------
// outAcc already holds x@Ws + bs (root skip), so segment_sum lands on top of it.
__global__ __launch_bounds__(256) void edge_scatter(
    const int* __restrict__ ei, const float* __restrict__ score,
    const float* __restrict__ denom, const float* __restrict__ V,
    float* __restrict__ outAcc) {
  const int wid = (int)((blockIdx.x * 256 + threadIdx.x) >> 5);
  if (wid >= EE) return;
  const int lane = threadIdx.x & 31;
  const int src = ei[wid];
  const int dst = ei[EE + wid];
  const int hh  = lane >> 3;
  const float alpha = score[(size_t)wid * HH + hh] / denom[dst * HH + hh];
  const float4* vp = (const float4*)(V + (size_t)src * DD + lane * 8);
  float4 v0 = vp[0], v1 = vp[1];
  float* op = outAcc + (size_t)dst * DD + lane * 8;
  atomicAdd(op + 0, alpha * v0.x);
  atomicAdd(op + 1, alpha * v0.y);
  atomicAdd(op + 2, alpha * v0.z);
  atomicAdd(op + 3, alpha * v0.w);
  atomicAdd(op + 4, alpha * v1.x);
  atomicAdd(op + 5, alpha * v1.y);
  atomicAdd(op + 6, alpha * v1.z);
  atomicAdd(op + 7, alpha * v1.w);
}

// ---------------- BatchNorm column statistics ----------------
__global__ __launch_bounds__(256) void col_stats(
    const float* __restrict__ S, float* __restrict__ colsum,
    float* __restrict__ colsumsq) {
  const int c = threadIdx.x;
  float s = 0.f, ss = 0.f;
  for (int r = blockIdx.x; r < NN; r += gridDim.x) {
    const float v = S[(size_t)r * DD + c];
    s += v; ss += v * v;
  }
  atomicAdd(colsum + c, s);
  atomicAdd(colsumsq + c, ss);
}

// ---------------- BN + ReLU, fused with bf16 re-quantization for next layer ----------------
__global__ __launch_bounds__(256) void bn_relu_cvt(
    const float* __restrict__ S, const float* __restrict__ colsum,
    const float* __restrict__ colsumsq, const float* __restrict__ gamma,
    const float* __restrict__ beta, bf16* __restrict__ Xb) {
  const int t = blockIdx.x * 256 + threadIdx.x;   // N*D threads
  const int c = t & (DD - 1);
  const float inv_n = 1.0f / (float)NN;
  const float mean = colsum[c] * inv_n;
  const float var  = colsumsq[c] * inv_n - mean * mean;
  const float y = gamma[c] * (S[t] - mean) * rsqrtf(var + 1e-5f) + beta[c];
  Xb[t] = (bf16)fmaxf(y, 0.f);
}

// ---------------- workspace layout (bytes, all 256B aligned) ----------------
#define OFF_XB     ((size_t)0)            // N*D bf16            = 10,240,000
#define OFF_WBT    ((size_t)10240000)     // 4*D*D bf16          =    524,288
#define OFF_Q      ((size_t)10764288)     // N*D f32             = 20,480,000
#define OFF_K      ((size_t)31244288)
#define OFF_V      ((size_t)51724288)
#define OFF_S      ((size_t)72204288)
#define OFF_SCORE  ((size_t)92684288)     // E*H f32             = 10,240,000
#define OFF_STATS  ((size_t)102924288)    // nodeMax,denom,colsum,colsumsq
#define STATS_U32  ((NN * HH * 2) + 512)  // 160,512 words

extern "C" void kernel_launch(void* const* d_in, const int* in_sizes, int n_in,
                              void* d_out, int out_size, void* d_ws, size_t ws_size,
                              hipStream_t stream) {
  const float* x  = (const float*)d_in[0];
  const int*   ei = (const int*)d_in[1];
  const float* Wq = (const float*)d_in[2];
  const float* bq = (const float*)d_in[3];
  const float* Wk = (const float*)d_in[4];
  const float* bk = (const float*)d_in[5];
  const float* Wv = (const float*)d_in[6];
  const float* bv = (const float*)d_in[7];
  const float* Ws = (const float*)d_in[8];
  const float* bs = (const float*)d_in[9];
  const float* bn_g = (const float*)d_in[10];
  const float* bn_b = (const float*)d_in[11];
  float* out = (float*)d_out;

  char* ws = (char*)d_ws;
  bf16*     Xb      = (bf16*)(ws + OFF_XB);
  bf16*     WbT     = (bf16*)(ws + OFF_WBT);
  float*    Qb      = (float*)(ws + OFF_Q);
  float*    Kb      = (float*)(ws + OFF_K);
  float*    Vb      = (float*)(ws + OFF_V);
  float*    Sb      = (float*)(ws + OFF_S);
  float*    score   = (float*)(ws + OFF_SCORE);
  unsigned* stats   = (unsigned*)(ws + OFF_STATS);
  unsigned* nodeMax = stats;
  float*    denom   = (float*)(stats + NN * HH);
  float*    colsum  = (float*)(stats + NN * HH * 2);
  float*    colsqs  = (float*)(stats + NN * HH * 2 + 256);

  const int nd_blocks = (NN * DD) / 256;      // 20000
  const int ew_blocks = EE / 8;               // 80000 (wave per edge)
  const int eh_blocks = (EE * HH) / 256;      // 10000

  // initial f32 -> bf16 feature conversion
  cvt_x_bf16<<<nd_blocks, 256, 0, stream>>>(x, Xb);

  for (int l = 0; l < LL; ++l) {
    // per-layer weight convert + transpose to bf16
    cvt_w_bf16<<<(4 * DD * DD) / 256, 256, 0, stream>>>(
        Wq + (size_t)l * DD * DD, Wk + (size_t)l * DD * DD,
        Wv + (size_t)l * DD * DD, Ws + (size_t)l * DD * DD, WbT);

    float* Starget = (l == LL - 1) ? out : Sb;

    // Q,K,V and root-skip GEMMs via WMMA (LDS-staged weight panels)
    dim3 ggrid(157, 4, 4);
    gemm4_wmma<<<ggrid, 256, 0, stream>>>(Xb, WbT,
        bq + (size_t)l * DD, bk + (size_t)l * DD,
        bv + (size_t)l * DD, bs + (size_t)l * DD,
        Qb, Kb, Vb, Starget);

    // clear per-layer reduction state (nodeMax/denom/colsum/colsumsq)
    zero_u32<<<(STATS_U32 + 255) / 256, 256, 0, stream>>>(stats, STATS_U32);

    // attention: scores + segment max, exp + segment sum, weighted scatter
    edge_scores<<<ew_blocks, 256, 0, stream>>>(ei, Qb, Kb, score, nodeMax);
    edge_exp<<<eh_blocks, 256, 0, stream>>>(ei, score, nodeMax, denom);
    edge_scatter<<<ew_blocks, 256, 0, stream>>>(ei, score, denom, Vb, Starget);

    if (l < LL - 1) {
      col_stats<<<160, 256, 0, stream>>>(Starget, colsum, colsqs);
      bn_relu_cvt<<<nd_blocks, 256, 0, stream>>>(Starget, colsum, colsqs,
          bn_g + (size_t)l * DD, bn_b + (size_t)l * DD, Xb);
    }
  }
}
